// TrajLSTM_57475252355308
// MI455X (gfx1250) — matrix operand order repaired and checked by reading, not compiled, not run on hardware
//
#include <hip/hip_runtime.h>
#include <hip/hip_bf16.h>
#include <math.h>

// Problem constants (match reference)
#define BATCH   16384
#define TSTEPS  50
#define FEAT    2
#define HID     64
#define HORIZON 50
#define NGATE   256   // 4*HID

typedef _Float16 v16h __attribute__((ext_vector_type(16)));
typedef _Float16 v8h  __attribute__((ext_vector_type(8)));
typedef float    v8f  __attribute__((ext_vector_type(8)));

static __device__ __forceinline__ float fast_tanh(float x) {
#if __has_builtin(__builtin_amdgcn_tanhf)
    return __builtin_amdgcn_tanhf(x);
#else
    return tanhf(x);
#endif
}
static __device__ __forceinline__ float fast_sigmoid(float x) {
    return 0.5f + 0.5f * fast_tanh(0.5f * x);
}

// ---------------------------------------------------------------------------
// Prep: shuffle W_hh (f32, [256][64] row-major) into f16 WMMA-B lane layout.
// B operand for output tile n (gate cols 16n..16n+15), K-chunk q (K 32q..32q+31):
//   lane l (<16):  half j -> (K = 32q + j (j<8) / 32q + 16 + j-8 (j>=8),      N = 16n + l)
//   lane l (>=16): half j -> (K = 32q + 8 + j (j<8) / 32q + 24 + j-8 (j>=8),  N = 16n + l-16)
// flat index = ((n*2+q)*32 + l)*16 + j   (16384 halfs = 32 KB)
// ---------------------------------------------------------------------------
__global__ void prep_whh_kernel(const float* __restrict__ whh,
                                _Float16* __restrict__ wsh) {
    int idx = blockIdx.x * blockDim.x + threadIdx.x;
    if (idx >= 16 * 2 * 32 * 16) return;
    int j = idx & 15;
    int l = (idx >> 4) & 31;
    int q = (idx >> 9) & 1;
    int n = idx >> 10;
    int K = 32 * q + ((j & 8) << 1) + 8 * (l >> 4) + (j & 7);
    int N = 16 * n + (l & 15);
    wsh[idx] = (_Float16)whh[N * HID + K];  // B[K][N] = W_hh[N][K]
}

// A-operand fragment from LDS h (f16, row-major [16][64]).
// lane l<16:  halfs 0-7 = K kb..kb+7, halfs 8-15 = K kb+16..kb+23 (kb = 32q)
// lane l>=16: same with kb = 32q + 8
static __device__ __forceinline__ v16h load_a(const _Float16* shh, int rowM, int kbase) {
    const v8h* p = (const v8h*)(shh + rowM * HID + kbase);
    v8h lo = p[0];
    v8h hi = p[2];
    v16h a;
#pragma unroll
    for (int j = 0; j < 8; ++j) { a[j] = lo[j]; a[8 + j] = hi[j]; }
    return a;
}

// One 16x16 gate tile with all loop-invariant operands already in registers:
// C = b + x*W_ih^T (seeded), then two K=32 WMMAs accumulate h*W_hh^T.
static __device__ __forceinline__ v8f gate_tile(const v16h a0, const v16h a1,
                                                const v16h b0, const v16h b1,
                                                float w0, float w1, float bb,
                                                const v8f xv0, const v8f xv1) {
    v8f c;
#pragma unroll
    for (int v = 0; v < 8; ++v)
        c[v] = fmaf(xv0[v], w0, fmaf(xv1[v], w1, bb));
    c = __builtin_amdgcn_wmma_f32_16x16x32_f16(false, a0, false, b0, (short)0, c, false, false);
    c = __builtin_amdgcn_wmma_f32_16x16x32_f16(false, a1, false, b1, (short)0, c, false, false);
    return c;
}

// ---------------------------------------------------------------------------
// Main kernel: 256 threads = 8 waves = 2 row-groups x 4 column-waves.
// Each column-wave owns one 16-wide gate column-group (8 WMMAs/step) and
// keeps ALL loop-invariant operands in VGPRs: 8 B fragments (64 regs) +
// per-gate scalar constants. Per-step LDS traffic is only the h exchange
// (2 reads + 1 write) and the x staging. h double-buffered; 2 barriers/step.
// ---------------------------------------------------------------------------
__global__ __launch_bounds__(256)
void traj_lstm_kernel(const float* __restrict__ history,
                      const float* __restrict__ W_ih,
                      const _Float16* __restrict__ whh_f16,  // prepped table in ws
                      const float* __restrict__ b_ih,
                      const float* __restrict__ b_hh,
                      const float* __restrict__ W_dec,
                      const float* __restrict__ b_dec,
                      float* __restrict__ out) {
    __shared__ v16h     s_whh[16 * 2 * 32];        // 32 KB, B operands (staging)
    __shared__ _Float16 s_h[2 * 2 * 16 * HID];     // [rowgroup][buf][16][64] f16
    __shared__ float    s_x[2 * 16 * FEAT];        // per-rowgroup x
    __shared__ float    s_wih[NGATE * FEAT];
    __shared__ float    s_b[NGATE];
    __shared__ float    s_wdec[FEAT * HID];
    __shared__ float    s_bdec[FEAT];

    const int tid  = threadIdx.x;
    const int wv   = tid >> 5;
    const int lane = tid & 31;
    const int lcol = lane & 15;   // N within tile / staging row
    const int lhi  = lane >> 4;   // 0: M 0-7 | 1: M 8-15  (C/D layout halves)
    const int grp  = wv >> 2;     // row-group within WG (0/1)
    const int cg   = wv & 3;      // column-group j (gate cols 16cg.. per gate)

    const int rowBase = (blockIdx.x * 2 + grp) * 16;
    float* myx = s_x + grp * (16 * FEAT);

    // ---- stage weights to LDS (cooperative, all 256 threads) ----
    {
        const uint4* src = (const uint4*)whh_f16;   // 32 KB = 2048 uint4
        uint4* dst = (uint4*)s_whh;
#pragma unroll
        for (int i = tid; i < 2048; i += 256) dst[i] = src[i];
        for (int i = tid; i < NGATE * FEAT; i += 256) s_wih[i] = W_ih[i];
        for (int i = tid; i < NGATE; i += 256) s_b[i] = b_ih[i] + b_hh[i];
        for (int i = tid; i < FEAT * HID; i += 256) s_wdec[i] = W_dec[i];
        if (tid < FEAT) s_bdec[tid] = b_dec[tid];
    }
    // zero h buffer 0 of this row-group (4 waves = 128 lanes cooperate)
    {
        _Float16* h0 = s_h + (grp * 2 + 0) * (16 * HID);
        int gtid = cg * 32 + lane;
        for (int i = gtid; i < 16 * HID; i += 128) h0[i] = (_Float16)0.0f;
    }
    // stage x for step 0
    if (cg == 0 && lane < 16) {
        const float* hp = history + (size_t)(rowBase + lane) * (TSTEPS * FEAT);
        myx[lane * 2 + 0] = hp[0];
        myx[lane * 2 + 1] = hp[1];
    }
    __syncthreads();

    // ---- pin ALL loop-invariant operands in registers ----
    // 8 WMMA B fragments for this wave's 4 gate tiles (i,f,g,o) x 2 K-chunks
    v16h bB[4][2];
#pragma unroll
    for (int g = 0; g < 4; ++g) {
        int n = cg + 4 * g;
        bB[g][0] = s_whh[(n * 2 + 0) * 32 + lane];
        bB[g][1] = s_whh[(n * 2 + 1) * 32 + lane];
    }
    // per-gate scalar constants for this lane's column
    float cw0[4], cw1[4], cbb[4];
#pragma unroll
    for (int g = 0; g < 4; ++g) {
        int col = (cg + 4 * g) * 16 + lcol;
        cw0[g] = s_wih[col * 2 + 0];
        cw1[g] = s_wih[col * 2 + 1];
        cbb[g] = s_b[col];
    }

    // cell state, D-tile layout: cst[v] = c[M = v + 8*lhi][16*cg + lcol]
    float cst[8];
#pragma unroll
    for (int v = 0; v < 8; ++v) cst[v] = 0.0f;

    int cur = 0;
    for (int t = 0; t < TSTEPS + HORIZON; ++t) {
        _Float16* hcur = s_h + (grp * 2 + cur) * (16 * HID);
        _Float16* hnxt = s_h + (grp * 2 + (cur ^ 1)) * (16 * HID);

        // ---- phase A: gates for this wave's column-group ----
        v8f xv0, xv1;
#pragma unroll
        for (int v = 0; v < 8; ++v) {
            int M = v + 8 * lhi;
            xv0[v] = myx[M * 2 + 0];
            xv1[v] = myx[M * 2 + 1];
        }
        const v16h a0 = load_a(hcur, lcol, 0  + 8 * lhi);   // K 0..31 chunk
        const v16h a1 = load_a(hcur, lcol, 32 + 8 * lhi);   // K 32..63 chunk

        v8f gi = gate_tile(a0, a1, bB[0][0], bB[0][1], cw0[0], cw1[0], cbb[0], xv0, xv1);
        v8f gf = gate_tile(a0, a1, bB[1][0], bB[1][1], cw0[1], cw1[1], cbb[1], xv0, xv1);
        v8f gg = gate_tile(a0, a1, bB[2][0], bB[2][1], cw0[2], cw1[2], cbb[2], xv0, xv1);
        v8f go = gate_tile(a0, a1, bB[3][0], bB[3][1], cw0[3], cw1[3], cbb[3], xv0, xv1);

#pragma unroll
        for (int v = 0; v < 8; ++v) {
            float i_s = fast_sigmoid(gi[v]);
            float f_s = fast_sigmoid(gf[v]);
            float g_t = fast_tanh(gg[v]);
            float o_s = fast_sigmoid(go[v]);
            float cn  = fmaf(f_s, cst[v], i_s * g_t);
            cst[v]    = cn;
            float hn  = o_s * fast_tanh(cn);
            int M     = v + 8 * lhi;
            int hcol  = cg * 16 + lcol;
            hnxt[M * HID + hcol] = (_Float16)hn;
        }
        __syncthreads();   // h[nxt] complete; phase-A reads of myx done

        // ---- phase B: one wave per row-group prepares x for step t+1 ----
        if (cg == 0) {
            if (t >= TSTEPS) {
                // decoder projection: 16 rows x 2 features across 32 lanes
                const int row = lcol;
                const int f   = lhi;
                float acc = s_bdec[f];
                const v8h* hr = (const v8h*)(hnxt + row * HID);
                const float* wd = s_wdec + f * HID;
#pragma unroll
                for (int c8 = 0; c8 < 8; ++c8) {
                    v8h hv = hr[c8];
#pragma unroll
                    for (int k = 0; k < 8; ++k)
                        acc = fmaf((float)hv[k], wd[c8 * 8 + k], acc);
                }
                size_t o = (size_t)(rowBase + row) * (HORIZON * FEAT)
                         + (size_t)(t - TSTEPS) * FEAT + f;
                out[o] = acc;
                myx[row * 2 + f] = acc;          // feedback x for next step
            } else if (t + 1 < TSTEPS) {
                if (lane < 16) {                  // prefetch next encoder x
                    const float* hp = history + (size_t)(rowBase + lane) * (TSTEPS * FEAT)
                                              + (size_t)(t + 1) * FEAT;
                    myx[lane * 2 + 0] = hp[0];
                    myx[lane * 2 + 1] = hp[1];
                }
            }
            // t == TSTEPS-1: keep myx == history[:, T-1, :] == x0 for decoder
        }
        __syncthreads();   // myx ready for step t+1
        cur ^= 1;
    }
}

extern "C" void kernel_launch(void* const* d_in, const int* in_sizes, int n_in,
                              void* d_out, int out_size, void* d_ws, size_t ws_size,
                              hipStream_t stream) {
    const float* history = (const float*)d_in[0];
    const float* W_ih    = (const float*)d_in[1];
    const float* W_hh    = (const float*)d_in[2];
    const float* b_ih    = (const float*)d_in[3];
    const float* b_hh    = (const float*)d_in[4];
    const float* W_dec   = (const float*)d_in[5];
    const float* b_dec   = (const float*)d_in[6];
    float* out = (float*)d_out;

    _Float16* whh_f16 = (_Float16*)d_ws;   // 32 KB table

    // 1) shuffle W_hh into WMMA-B f16 layout
    prep_whh_kernel<<<64, 256, 0, stream>>>(W_hh, whh_f16);

    // 2) recurrence: 16384 rows / (16 rows * 2 row-groups per WG) = 512 WGs,
    //    4 column-waves per row-group -> 4096 waves machine-wide.
    traj_lstm_kernel<<<512, 256, 0, stream>>>(history, W_ih, whh_f16,
                                              b_ih, b_hh, W_dec, b_dec, out);
}
